// MP_Model_2465311227921
// MI455X (gfx1250) — matrix operand order repaired and checked by reading, not compiled, not run on hardware
//
#include <hip/hip_runtime.h>
#include <hip/hip_bf16.h>

#define BATCH   32
#define DMODEL  16384
#define NATOM   4096
#define MP_ITERS 10
#define INV_TEMP 1000.0f

typedef __bf16 bf16_t;
typedef __attribute__((ext_vector_type(16))) __bf16 v16bf;
typedef __attribute__((ext_vector_type(8)))  __bf16 v8bf;
typedef __attribute__((ext_vector_type(8)))  float  v8f;

#define WMMA_BF16(a, b, c) \
    __builtin_amdgcn_wmma_f32_16x16x32_bf16(false, (a), false, (b), (short)0, (c), false, false)

static __device__ __forceinline__ v16bf cat16(v8bf lo, v8bf hi) {
    return __builtin_shufflevector(lo, hi, 0,1,2,3,4,5,6,7,8,9,10,11,12,13,14,15);
}

// float -> bf16 with round-to-nearest-even (pure bit ops: always compiles)
static __device__ __forceinline__ bf16_t f2bf(float f) {
    unsigned u = __float_as_uint(f);
    u += 0x7fffu + ((u >> 16) & 1u);
    unsigned short h = (unsigned short)(u >> 16);
    bf16_t r;
    __builtin_memcpy(&r, &h, 2);
    return r;
}

// 16-bit A-matrix 16x32: lane<16 holds K=k0+0..15;
// lane>=16 holds K=k0+16..31; row M = lane&15 (+ mbase).
static __device__ __forceinline__ v16bf load_a_tile(const bf16_t* __restrict__ X, int ld,
                                                    int mbase, int k0, int lane) {
    int m  = mbase + (lane & 15);
    int kh = (lane >> 4) * 16;
    const v8bf* p = (const v8bf*)(X + (size_t)m * ld + k0 + kh);
    return cat16(p[0], p[1]);   // +0..7  and  +8..15 (elements)
}

// 16-bit B-matrix 32x16: lane = column n; lane<16 holds K=k0+0..15,
// lane>=16 holds K=k0+16..31. B[k][n] = Y[rbase+n][k0+k] (contiguous in Y rows).
static __device__ __forceinline__ v16bf load_b_tile(const bf16_t* __restrict__ Y, int ld,
                                                    int rbase, int k0, int lane) {
    int n  = lane & 15;
    int kh = (lane >> 4) * 16;
    const v8bf* p = (const v8bf*)(Y + (size_t)(rbase + n) * ld + k0 + kh);
    return cat16(p[0], p[1]);
}

// Non-temporal variant (keep the other dictionary copy resident in L2)
static __device__ __forceinline__ v16bf load_b_tile_nt(const bf16_t* __restrict__ Y, int ld,
                                                       int rbase, int k0, int lane) {
    int n  = lane & 15;
    int kh = (lane >> 4) * 16;
    const v8bf* p = (const v8bf*)(Y + (size_t)(rbase + n) * ld + k0 + kh);
    v8bf lo = __builtin_nontemporal_load(p);
    v8bf hi = __builtin_nontemporal_load(p + 1);
    return cat16(lo, hi);
}

// 32-bit C/D 16x16: VGPR r -> (M = r + 8*(lane>=16), N = lane&15)
static __device__ __forceinline__ void store_c_tile(float* __restrict__ out, int ldc,
                                                    int mbase, int nbase, v8f c, int lane) {
    int n  = nbase + (lane & 15);
    int mo = mbase + (lane >> 4) * 8;
#pragma unroll
    for (int r = 0; r < 8; ++r)
        out[(size_t)(mo + r) * ldc + n] = c[r];
}

// ---------------------------------------------------------------------------
// Kernel 1: dict fp32 -> bf16 row-major copy + bf16 transposed copy
// grid (DMODEL/32, NATOM/32), block (32,8)
// ---------------------------------------------------------------------------
__global__ __launch_bounds__(256) void prep_kernel(const float* __restrict__ dict,
                                                   bf16_t* __restrict__ db,
                                                   bf16_t* __restrict__ dT) {
    __shared__ float t[32][33];
    int cx = blockIdx.x * 32;     // column base (D dim)
    int ry = blockIdx.y * 32;     // row base (atom dim)
    int tx = threadIdx.x;
    int ty = threadIdx.y;
#pragma unroll
    for (int j = 0; j < 4; ++j) {
        int r = ry + ty + j * 8;
        float v = dict[(size_t)r * DMODEL + cx + tx];
        t[ty + j * 8][tx] = v;
        db[(size_t)r * DMODEL + cx + tx] = f2bf(v);
    }
    __syncthreads();
#pragma unroll
    for (int j = 0; j < 4; ++j) {
        int c = cx + ty + j * 8;
        dT[(size_t)c * NATOM + ry + tx] = f2bf(t[tx][ty + j * 8]);
    }
}

// ---------------------------------------------------------------------------
// Kernel 2: residual init (fp32 + bf16 mirror). grid 2048 x 256
// ---------------------------------------------------------------------------
__global__ __launch_bounds__(256) void init_kernel(const float* __restrict__ x,
                                                   float* __restrict__ res,
                                                   bf16_t* __restrict__ resb) {
    int i = blockIdx.x * 256 + threadIdx.x;
    float v = x[i];
    res[i]  = v;
    resb[i] = f2bf(v);
}

// ---------------------------------------------------------------------------
// Kernel 3: scores (partial, split-K over 8 waves).  grid 64 x 256.
// Wave w: K slice [w*2048, w*2048+2048), 64 atoms (4 N-tiles), 2 M-tiles.
// Writes spart[w][32][4096].
// ---------------------------------------------------------------------------
__global__ __launch_bounds__(256) void scores_kernel(const bf16_t* __restrict__ rbf,
                                                     const bf16_t* __restrict__ dictb,
                                                     float* __restrict__ spart) {
    int lane = threadIdx.x & 31;
    int wave = threadIdx.x >> 5;
    int atomBase = blockIdx.x * 64;
    int kbase = wave * (DMODEL / 8);

    v8f z = {0.f, 0.f, 0.f, 0.f, 0.f, 0.f, 0.f, 0.f};
    v8f acc[2][4];
#pragma unroll
    for (int m = 0; m < 2; ++m)
#pragma unroll
        for (int n = 0; n < 4; ++n) acc[m][n] = z;

    for (int ks = 0; ks < DMODEL / 8; ks += 32) {
        int k0 = kbase + ks;
        v16bf a0 = load_a_tile(rbf, DMODEL, 0, k0, lane);
        v16bf a1 = load_a_tile(rbf, DMODEL, 16, k0, lane);
        __builtin_prefetch(dictb + (size_t)(atomBase + (lane & 15)) * DMODEL + k0 + 64, 0, 0);
#pragma unroll
        for (int nt = 0; nt < 4; ++nt) {
            v16bf bt = load_b_tile(dictb, DMODEL, atomBase + nt * 16, k0, lane);
            acc[0][nt] = WMMA_BF16(a0, bt, acc[0][nt]);
            acc[1][nt] = WMMA_BF16(a1, bt, acc[1][nt]);
        }
    }
    float* out = spart + (size_t)wave * BATCH * NATOM;
#pragma unroll
    for (int m = 0; m < 2; ++m)
#pragma unroll
        for (int nt = 0; nt < 4; ++nt)
            store_c_tile(out, NATOM, m * 16, atomBase + nt * 16, acc[m][nt], lane);
}

// ---------------------------------------------------------------------------
// Kernel 4: sum split-K partials + softmax(x*1000) -> bf16 weights.
// grid 32 (one block per batch row) x 256.
// ---------------------------------------------------------------------------
__global__ __launch_bounds__(256) void softmax_kernel(const float* __restrict__ spart,
                                                      bf16_t* __restrict__ wts) {
    __shared__ float red[256];
    int b = blockIdx.x, tid = threadIdx.x;
    float loc[16];
    float mx = -3.4e38f;
#pragma unroll
    for (int i = 0; i < 16; ++i) {
        int a = tid + i * 256;
        float s = 0.f;
#pragma unroll
        for (int sl = 0; sl < 8; ++sl)
            s += spart[((size_t)sl * BATCH + b) * NATOM + a];
        s *= INV_TEMP;
        loc[i] = s;
        mx = fmaxf(mx, s);
    }
    red[tid] = mx; __syncthreads();
    for (int off = 128; off > 0; off >>= 1) {
        if (tid < off) red[tid] = fmaxf(red[tid], red[tid + off]);
        __syncthreads();
    }
    mx = red[0];
    __syncthreads();
    float sum = 0.f;
#pragma unroll
    for (int i = 0; i < 16; ++i) { float e = expf(loc[i] - mx); loc[i] = e; sum += e; }
    red[tid] = sum; __syncthreads();
    for (int off = 128; off > 0; off >>= 1) {
        if (tid < off) red[tid] += red[tid + off];
        __syncthreads();
    }
    float inv = 1.0f / red[0];
#pragma unroll
    for (int i = 0; i < 16; ++i)
        wts[(size_t)b * NATOM + tid + i * 256] = f2bf(loc[i] * inv);
}

// ---------------------------------------------------------------------------
// Kernel 5: recon partial = W @ D via Dᵀ (bf16, non-temporal). grid 64 x 256.
// WG covers 256 output cols; 8 waves = 4 col-groups x 2 K-slices (2048 atoms).
// ---------------------------------------------------------------------------
__global__ __launch_bounds__(256) void recon_kernel(const bf16_t* __restrict__ wts,
                                                    const bf16_t* __restrict__ dictT,
                                                    float* __restrict__ rpart) {
    int lane = threadIdx.x & 31;
    int wave = threadIdx.x >> 5;
    int cg  = wave >> 1;
    int ksl = wave & 1;
    int nbase = blockIdx.x * 256 + cg * 64;
    int kbase = ksl * (NATOM / 2);

    v8f z = {0.f, 0.f, 0.f, 0.f, 0.f, 0.f, 0.f, 0.f};
    v8f acc[2][4];
#pragma unroll
    for (int m = 0; m < 2; ++m)
#pragma unroll
        for (int n = 0; n < 4; ++n) acc[m][n] = z;

    for (int ks = 0; ks < NATOM / 2; ks += 32) {
        int k0 = kbase + ks;
        v16bf a0 = load_a_tile(wts, NATOM, 0, k0, lane);
        v16bf a1 = load_a_tile(wts, NATOM, 16, k0, lane);
#pragma unroll
        for (int nt = 0; nt < 4; ++nt) {
            v16bf bt = load_b_tile_nt(dictT, NATOM, nbase + nt * 16, k0, lane);
            acc[0][nt] = WMMA_BF16(a0, bt, acc[0][nt]);
            acc[1][nt] = WMMA_BF16(a1, bt, acc[1][nt]);
        }
    }
    float* out = rpart + (size_t)ksl * BATCH * DMODEL;
#pragma unroll
    for (int m = 0; m < 2; ++m)
#pragma unroll
        for (int nt = 0; nt < 4; ++nt)
            store_c_tile(out, DMODEL, m * 16, nbase + nt * 16, acc[m][nt], lane);
}

// ---------------------------------------------------------------------------
// Kernel 6: residual -= (p0 + p1); refresh bf16 residual. grid 2048 x 256
// ---------------------------------------------------------------------------
__global__ __launch_bounds__(256) void update_kernel(float* __restrict__ res,
                                                     bf16_t* __restrict__ resb,
                                                     const float* __restrict__ rpart) {
    int i = blockIdx.x * 256 + threadIdx.x;
    float r = res[i] - (rpart[i] + rpart[i + BATCH * DMODEL]);
    res[i]  = r;
    resb[i] = f2bf(r);
}

// ---------------------------------------------------------------------------
// Kernel 7: copy residual to output + deterministic partial sum of squares.
// grid 512 x 256 (1024 elements per block).
// ---------------------------------------------------------------------------
__global__ __launch_bounds__(256) void loss_partial_kernel(const float* __restrict__ res,
                                                           float* __restrict__ outRes,
                                                           float* __restrict__ partial) {
    __shared__ float sh[256];
    int base = blockIdx.x * 1024;
    float s = 0.f;
#pragma unroll
    for (int j = 0; j < 4; ++j) {
        int i = base + threadIdx.x + j * 256;
        float v = res[i];
        outRes[i] = v;
        s += v * v;
    }
    sh[threadIdx.x] = s; __syncthreads();
    for (int off = 128; off > 0; off >>= 1) {
        if (threadIdx.x < off) sh[threadIdx.x] += sh[threadIdx.x + off];
        __syncthreads();
    }
    if (threadIdx.x == 0) partial[blockIdx.x] = sh[0];
}

__global__ __launch_bounds__(256) void loss_final_kernel(const float* __restrict__ partial,
                                                         float* __restrict__ out0) {
    __shared__ float sh[256];
    int tid = threadIdx.x;
    sh[tid] = partial[tid] + partial[tid + 256];
    __syncthreads();
    for (int off = 128; off > 0; off >>= 1) {
        if (tid < off) sh[tid] += sh[tid + off];
        __syncthreads();
    }
    if (tid == 0) out0[0] = sh[0] / (float)(BATCH * DMODEL);
}

// ---------------------------------------------------------------------------
extern "C" void kernel_launch(void* const* d_in, const int* in_sizes, int n_in,
                              void* d_out, int out_size, void* d_ws, size_t ws_size,
                              hipStream_t stream) {
    const float* x_eeg = (const float*)d_in[0];   // [32, 64, 256] fp32
    const float* dict  = (const float*)d_in[1];   // [4096, 16384] fp32
    float* out = (float*)d_out;                   // [0]=loss, [1..524288]=residual

    char* ws = (char*)d_ws;
    const size_t SZ_DICT16 = (size_t)NATOM * DMODEL * sizeof(bf16_t);   // 128 MB
    bf16_t* dictb = (bf16_t*)(ws);
    bf16_t* dictT = (bf16_t*)(ws + SZ_DICT16);
    size_t off = 2 * SZ_DICT16;
    float*  res   = (float*) (ws + off); off += (size_t)BATCH * DMODEL * sizeof(float);
    bf16_t* resb  = (bf16_t*)(ws + off); off += (size_t)BATCH * DMODEL * sizeof(bf16_t);
    float*  spart = (float*) (ws + off); off += (size_t)8 * BATCH * NATOM * sizeof(float);
    bf16_t* wts   = (bf16_t*)(ws + off); off += (size_t)BATCH * NATOM * sizeof(bf16_t);
    float*  rpart = (float*) (ws + off); off += (size_t)2 * BATCH * DMODEL * sizeof(float);
    float*  partial = (float*)(ws + off);

    // one-time: build bf16 dictionary (row-major) + bf16 transposed dictionary
    prep_kernel<<<dim3(DMODEL / 32, NATOM / 32), dim3(32, 8), 0, stream>>>(dict, dictb, dictT);
    init_kernel<<<2048, 256, 0, stream>>>(x_eeg, res, resb);

    for (int it = 0; it < MP_ITERS; ++it) {
        scores_kernel <<<64, 256, 0, stream>>>(resb, dictb, spart);
        softmax_kernel<<<32, 256, 0, stream>>>(spart, wts);
        recon_kernel  <<<64, 256, 0, stream>>>(wts, dictT, rpart);
        update_kernel <<<2048, 256, 0, stream>>>(res, resb, rpart);
    }

    loss_partial_kernel<<<512, 256, 0, stream>>>(res, out + 1, partial);
    loss_final_kernel  <<<1, 256, 0, stream>>>(partial, out);
}
